// Net2_43207370997828
// MI455X (gfx1250) — matrix operand order repaired and checked by reading, not compiled, not run on hardware
//
#include <hip/hip_runtime.h>
#include <math.h>

#define N_NODES 100000
#define E_EDGES 1600000
#define F_IN    256
#define F_OUT   128

typedef __attribute__((ext_vector_type(16))) __bf16 v16bf;
typedef __attribute__((ext_vector_type(8)))  float  v8f;

// ---------------------------------------------------------------------------
// 1) degree init: deg[i] = 1 (self loop)
__global__ void k_init_deg(float* __restrict__ deg) {
    int i = blockIdx.x * blockDim.x + threadIdx.x;
    if (i < N_NODES) deg[i] = 1.0f;
}

// 2) deg[dst[e]] += 1
__global__ void k_count_deg(const int* __restrict__ dst, float* __restrict__ deg) {
    int e = blockIdx.x * blockDim.x + threadIdx.x;
    if (e < E_EDGES) atomicAdd(&deg[dst[e]], 1.0f);
}

// 3) dinv[i] = rsqrt(deg[i])   (deg >= 1 always, self loop)
__global__ void k_rsqrt(float* __restrict__ deg) {
    int i = blockIdx.x * blockDim.x + threadIdx.x;
    if (i < N_NODES) deg[i] = rsqrtf(deg[i]);
}

// ---------------------------------------------------------------------------
// 4) x = feature @ W  via v_wmma_f32_16x16x32_bf16.
//    Block = 256 thr = 8 waves; each wave owns one 16-row M tile and all
//    8 N tiles (F_OUT=128). W^T staged in LDS as bf16 (64 KB).
__global__ void __launch_bounds__(256)
gcn_gemm_wmma(const float* __restrict__ feat, const float* __restrict__ W,
              float* __restrict__ x, int nrows) {
    __shared__ __align__(32) __bf16 sWt[F_OUT * F_IN];   // [col][k], 64 KB

    const int tid = threadIdx.x;
    // cooperative transpose + f32->bf16 of W into LDS
    for (int i = tid; i < F_OUT * F_IN; i += 256) {
        int c = i >> 8;      // 0..127  (col)
        int k = i & 255;     // 0..255  (k)
        sWt[i] = (__bf16)W[k * F_OUT + c];
    }
    __syncthreads();

    const int lane = tid & 31;
    const int wave = tid >> 5;
    const int half = lane >> 4;   // 0: lanes 0-15, 1: lanes 16-31
    const int l16  = lane & 15;

    const int rowBase = (blockIdx.x * 8 + wave) * 16;
    int ar = rowBase + l16;
    if (ar >= nrows) ar = nrows - 1;           // clamp loads; stores guarded
    const float* arow = feat + (size_t)ar * F_IN;

    v8f acc[8] = {};

    for (int k0 = 0; k0 < F_IN; k0 += 32) {
        // A fragment: this code's k order (same as B) -> lane supplies
        //   slots 0..15 : K = k0 + half*16 + s       (contiguous)
        const float* ap = arow + k0 + half * 16;
        float4 a0 = *(const float4*)(ap);
        float4 a1 = *(const float4*)(ap + 4);
        float4 a2 = *(const float4*)(ap + 8);
        float4 a3 = *(const float4*)(ap + 12);
        v16bf a;
        a[0]  = (__bf16)a0.x; a[1]  = (__bf16)a0.y; a[2]  = (__bf16)a0.z; a[3]  = (__bf16)a0.w;
        a[4]  = (__bf16)a1.x; a[5]  = (__bf16)a1.y; a[6]  = (__bf16)a1.z; a[7]  = (__bf16)a1.w;
        a[8]  = (__bf16)a2.x; a[9]  = (__bf16)a2.y; a[10] = (__bf16)a2.z; a[11] = (__bf16)a2.w;
        a[12] = (__bf16)a3.x; a[13] = (__bf16)a3.y; a[14] = (__bf16)a3.z; a[15] = (__bf16)a3.w;

#pragma unroll
        for (int nt = 0; nt < 8; ++nt) {
            // B fragment: same k order as A -> lane l16 = N, slot s gives
            // K = k0 + half*16 + s : one contiguous 32 B LDS read
            const int col = nt * 16 + l16;
            v16bf b = *(const v16bf*)&sWt[col * F_IN + k0 + half * 16];
            acc[nt] = __builtin_amdgcn_wmma_f32_16x16x32_bf16(
                false, a, false, b, (short)0, acc[nt], false, false);
        }
    }

    // C/D layout: VGPR r, lanes 0-15 -> M=r ; lanes 16-31 -> M=8+r ; N=l16
#pragma unroll
    for (int r = 0; r < 8; ++r) {
        const int row = rowBase + half * 8 + r;
        if (row < nrows) {
            float* orow = x + (size_t)row * F_OUT;
#pragma unroll
            for (int nt = 0; nt < 8; ++nt)
                orow[nt * 16 + l16] = acc[nt][r];
        }
    }
}

// ---------------------------------------------------------------------------
// 5) self-loop message: agg[n] = x[n] * dinv[n]^2  (initializes d_out fully)
__global__ void k_selfloop(const float* __restrict__ x, const float* __restrict__ dinv,
                           float* __restrict__ agg) {
    int t = blockIdx.x * blockDim.x + threadIdx.x;     // (node, lane-of-32)
    int node = t >> 5, lane = t & 31;
    if (node >= N_NODES) return;
    float s = dinv[node]; s *= s;
    const float4 v = ((const float4*)(x + (size_t)node * F_OUT))[lane];
    float4 o; o.x = v.x * s; o.y = v.y * s; o.z = v.z * s; o.w = v.w * s;
    ((float4*)(agg + (size_t)node * F_OUT))[lane] = o;
}

__global__ void k_zero128(float* __restrict__ p) { p[threadIdx.x] = 0.0f; }

// 6) edge scatter: one wave32 per edge, one float4 per lane (128 floats)
__global__ void __launch_bounds__(256)
k_scatter(const float* __restrict__ x, const float* __restrict__ dinv,
          const int* __restrict__ src, const int* __restrict__ dst,
          float* __restrict__ agg, int nedges) {
    int gwave = blockIdx.x * 8 + (threadIdx.x >> 5);
    int lane  = threadIdx.x & 31;
    if (gwave >= nedges) return;
    const int r = src[gwave];
    const int c = dst[gwave];
    const float nrm = dinv[r] * dinv[c];
    const float4 v = ((const float4*)(x + (size_t)r * F_OUT))[lane];
    float* o = agg + (size_t)c * F_OUT + lane * 4;
    atomicAdd(o + 0, v.x * nrm);
    atomicAdd(o + 1, v.y * nrm);
    atomicAdd(o + 2, v.z * nrm);
    atomicAdd(o + 3, v.w * nrm);
}

// 7) x_out = relu(agg + b) in place; accumulate column sums (wave -> LDS -> 1
//    global atomic per block per feature)
__global__ void __launch_bounds__(256)
k_finalize(float* __restrict__ agg, const float* __restrict__ bias,
           float* __restrict__ colsum, int nrows) {
    __shared__ float ls[F_OUT];
    const int tid  = threadIdx.x;
    if (tid < F_OUT) ls[tid] = 0.0f;
    __syncthreads();

    const int lane = tid & 31;
    const int wave = tid >> 5;
    const float4 bb = ((const float4*)bias)[lane];

    float sx = 0.f, sy = 0.f, sz = 0.f, sw = 0.f;
#pragma unroll
    for (int i = 0; i < 8; ++i) {
        const int node = blockIdx.x * 64 + wave * 8 + i;
        if (node < nrows) {
            float4* p = (float4*)(agg + (size_t)node * F_OUT) + lane;
            float4 v = *p;
            v.x = fmaxf(v.x + bb.x, 0.f);
            v.y = fmaxf(v.y + bb.y, 0.f);
            v.z = fmaxf(v.z + bb.z, 0.f);
            v.w = fmaxf(v.w + bb.w, 0.f);
            *p = v;
            sx += v.x; sy += v.y; sz += v.z; sw += v.w;
        }
    }
    atomicAdd(&ls[lane * 4 + 0], sx);
    atomicAdd(&ls[lane * 4 + 1], sy);
    atomicAdd(&ls[lane * 4 + 2], sz);
    atomicAdd(&ls[lane * 4 + 3], sw);
    __syncthreads();
    if (tid < F_OUT) atomicAdd(&colsum[tid], ls[tid]);
}

// 8) h = sigmoid(colsum / N)
__global__ void k_head(const float* __restrict__ colsum, float* __restrict__ h) {
    int f = threadIdx.x;
    float m = colsum[f] * (1.0f / (float)N_NODES);
    h[f] = 1.0f / (1.0f + expf(-m));
}

// ---------------------------------------------------------------------------
extern "C" void kernel_launch(void* const* d_in, const int* in_sizes, int n_in,
                              void* d_out, int out_size, void* d_ws, size_t ws_size,
                              hipStream_t stream) {
    const float* feat = (const float*)d_in[0];
    const int*   ei   = (const int*)d_in[1];
    const float* W    = (const float*)d_in[2];
    const float* bias = (const float*)d_in[3];
    const int* src = ei;             // edge_index[0]
    const int* dst = ei + E_EDGES;   // edge_index[1]

    float* out = (float*)d_out;      // [N*F_OUT] x_out, then [F_OUT] h

    float* x      = (float*)d_ws;                         // N*F_OUT f32
    float* dinv   = x + (size_t)N_NODES * F_OUT;          // N f32
    float* colsum = dinv + N_NODES;                       // F_OUT f32

    k_init_deg <<<(N_NODES + 255) / 256, 256, 0, stream>>>(dinv);
    k_count_deg<<<(E_EDGES + 255) / 256, 256, 0, stream>>>(dst, dinv);
    k_rsqrt    <<<(N_NODES + 255) / 256, 256, 0, stream>>>(dinv);

    gcn_gemm_wmma<<<(N_NODES + 127) / 128, 256, 0, stream>>>(feat, W, x, N_NODES);

    k_selfloop <<<(N_NODES * 32 + 255) / 256, 256, 0, stream>>>(x, dinv, out);
    k_zero128  <<<1, 128, 0, stream>>>(colsum);
    k_scatter  <<<(E_EDGES + 7) / 8, 256, 0, stream>>>(x, dinv, src, dst, out, E_EDGES);
    k_finalize <<<(N_NODES + 63) / 64, 256, 0, stream>>>(out, bias, colsum, N_NODES);
    k_head     <<<1, 128, 0, stream>>>(colsum, out + (size_t)N_NODES * F_OUT);
}
